// Attention_layer_true_i2mv_75222057222609
// MI455X (gfx1250) — matrix operand order repaired and checked by reading, not compiled, not run on hardware
//
#include <hip/hip_runtime.h>
#include <hip/hip_bf16.h>
#include <stdint.h>

// ---------------- problem constants (from reference) ----------------
#define I_DIM   128
#define T_DIM   128
#define QL      32
#define KL      32
#define HEADS   4
#define HD      16           // head dim
#define DM      64           // d_model
#define T_TILE  8            // t-values per block in main kernel
#define TILE_HALFS (HEADS * KL * HD)   // 2048 f16 elems = 4KB per K/V tile

typedef __attribute__((ext_vector_type(16))) _Float16 v16h;
typedef __attribute__((ext_vector_type(8)))  float    v8f;
typedef __attribute__((ext_vector_type(4)))  unsigned int u32x4;
typedef __attribute__((ext_vector_type(8)))  int          i32x8;
typedef __attribute__((ext_vector_type(4)))  int          i32x4;

__device__ __forceinline__ v8f wmma_f32_16x16x32(v16h a, v16h b, v8f c) {
  // (neg_a, A, neg_b, B, c_mod, C, reuse_a, reuse_b)
  return __builtin_amdgcn_wmma_f32_16x16x32_f16(false, a, false, b, (short)0, c,
                                                false, false);
}

// Raw LDS byte offset of a __shared__ object (generic ptr low 32 bits = LDS addr).
__device__ __forceinline__ unsigned lds_off(const void* p) {
  return (unsigned)(uintptr_t)p;
}

// Tensor Data Mover: flat 1-D copy of `nelem` f16 elements, global -> LDS.
// D# per CDNA5 ISA §8: group0 = count|lds_addr|global_addr|type=2,
// group1 = data_size=2B, tensor_dim0=tile_dim0=nelem, dim0_stride=nelem.
// This toolchain exposes the 6-arg builtin: (g0, g1, g2, g3, g4, cpol).
__device__ __forceinline__ void tdm_load_f16_1d(unsigned lds_byte_off,
                                                unsigned long long gaddr,
                                                unsigned nelem) {
  u32x4 g0;
  g0[0] = 1u;                                        // count=1, user mode
  g0[1] = lds_byte_off;                              // lds_addr
  g0[2] = (unsigned)gaddr;                           // global_addr[31:0]
  g0[3] = ((unsigned)(gaddr >> 32) & 0x01FFFFFFu)    // global_addr[56:32]
          | 0x80000000u;                             // type=2 (bits127:126=10b)
  i32x8 g1;
  g1[0] = 0x00010000;                                // data_size=1 (2 bytes)
  g1[1] = (int)(nelem << 16);                        // tensor_dim0[15:0]
  g1[2] = (int)((nelem >> 16) | (1u << 16));         // tensor_dim0[31:16] | tensor_dim1=1
  g1[3] = (int)(nelem << 16);                        // tile_dim0 (16-bit)
  g1[4] = 0;                                         // tile_dim1/2 unused
  g1[5] = (int)nelem;                                // tensor_dim0_stride[31:0]
  g1[6] = 0;
  g1[7] = 0;
  i32x4 gz4 = {0, 0, 0, 0};                          // groups 2/3: <=2D tensor
  i32x8 gz8 = {0, 0, 0, 0, 0, 0, 0, 0};              // trailing group (unused)
  __builtin_amdgcn_tensor_load_to_lds(g0, g1, gz4, gz4, gz8, 0);
}

// ---------------------------------------------------------------------------
// Kernel 1: per-head projections (f32 -> f16).
//   qn [((i*H+h)*QL+q)*HD+e] = sum_d query[i,q,h*16+d] * W_Q[e,d]
//   kn [((t*H+h)*KL+k)*HD+e] = sum_d key  [t,k,h*16+d] * W_K[e,d]
//   vnT[((t*H+h)*HD+e)*KL+k] = sum_d key  [t,k,h*16+d] * W_V[e,d]   (transposed!)
//   wfc = f16(W_fc)
// vnT is stored K-major so the P@V B-fragment (B[k][e], k varying per half)
// becomes a single contiguous 32-byte load.
// ---------------------------------------------------------------------------
__global__ __launch_bounds__(256)
void attn_proj_f16_kernel(const float* __restrict__ query,
                          const float* __restrict__ key,
                          const float* __restrict__ W_Q,
                          const float* __restrict__ W_K,
                          const float* __restrict__ W_V,
                          const float* __restrict__ W_fc,
                          _Float16* __restrict__ qn,
                          _Float16* __restrict__ kn,
                          _Float16* __restrict__ vnT,
                          _Float16* __restrict__ wfc) {
  const int NQ = I_DIM * HEADS * QL * HD;   // 262144 per tensor
  int idx = blockIdx.x * blockDim.x + threadIdx.x;

  if (idx < 3 * NQ) {
    const int which = idx / NQ;             // 0:qn 1:kn 2:vnT
    const int r     = idx - which * NQ;
    const int e = r & 15;
    const int q = (r >> 4) & (QL - 1);      // q (or k) row
    const int h = (r >> 9) & (HEADS - 1);
    const int i = r >> 11;                  // i (or t)
    const float* src = (which == 0 ? query : key) + ((size_t)i * QL + q) * DM + h * HD;
    const float* w   = (which == 0 ? W_Q : (which == 1 ? W_K : W_V)) + e * HD;
    float acc = 0.f;
#pragma unroll
    for (int d = 0; d < HD; ++d) acc += src[d] * w[d];
    if (which == 0)      qn[r] = (_Float16)acc;
    else if (which == 1) kn[r] = (_Float16)acc;
    else  vnT[(((size_t)i * HEADS + h) * HD + e) * KL + q] = (_Float16)acc;
  } else if (idx < 3 * NQ + DM * DM) {
    const int r = idx - 3 * NQ;
    wfc[r] = (_Float16)W_fc[r];
  }
}

// ---------------------------------------------------------------------------
// Kernel 2: main attention. 4 waves (= 4 heads) per block; block owns one i
// and T_TILE consecutive t. qn A-fragments + W_fc B-fragments are persistent
// in VGPRs. K/V tiles are double-buffered in LDS, filled by the Tensor Data
// Mover (tensor_load_to_lds, TENSORcnt) overlapping the WMMA/softmax work.
// ---------------------------------------------------------------------------
__global__ __launch_bounds__(128)
void attn_main_wmma_kernel(const _Float16* __restrict__ qn,
                           const _Float16* __restrict__ kn,
                           const _Float16* __restrict__ vnT,
                           const _Float16* __restrict__ wfc,
                           const unsigned char* __restrict__ kmask,
                           float* __restrict__ out) {
  __shared__ _Float16 kbuf[2][HEADS][KL][HD];   // 2 x 4KB  (K tiles)
  __shared__ _Float16 vbuf[2][HEADS][HD][KL];   // 2 x 4KB  (V^T tiles)
  __shared__ float    S[HEADS][QL][KL + 1];     // raw scores (padded rows)
  __shared__ _Float16 P[HEADS][QL][KL];         // softmax probs, f16
  __shared__ _Float16 VA[QL][DM];               // concat-head attn output, f16

  const int tid  = threadIdx.x;
  const int wave = tid >> 5;      // 0..3 : head id, also out n-tile id
  const int lane = tid & 31;
  const int ln   = lane & 15;
  const int hi   = lane >> 4;     // 0 or 1 (lane half-group)

  const int i  = blockIdx.x >> 4;                 // 128 i values
  const int t0 = (blockIdx.x & 15) * T_TILE;      // 16 t-groups of 8

  const unsigned long long kn_base = (unsigned long long)(uintptr_t)kn;
  const unsigned long long vt_base = (unsigned long long)(uintptr_t)vnT;

  // ---- persistent A fragments: qn for this head, 2 m-tiles (K padded 16->32)
  v16h aq[2];
#pragma unroll
  for (int mt = 0; mt < 2; ++mt) {
    const int m = mt * 16 + ln;
    const _Float16* src = qn + (((size_t)i * HEADS + wave) * QL + m) * HD + hi * 8;
    v16h a;
#pragma unroll
    for (int j = 0; j < 8; ++j) a[j] = src[j];         // k = hi*8 + j (k<16)
#pragma unroll
    for (int j = 8; j < 16; ++j) a[j] = (_Float16)0.f; // k>=16 zero-padded
    aq[mt] = a;
  }

  // ---- persistent B fragments of W_fc^T: n-tile = wave, 2 K-chunks of 32
  v16h bw[2];
#pragma unroll
  for (int kk = 0; kk < 2; ++kk) {
    const int e = wave * 16 + ln;
    const _Float16* src = wfc + e * DM + kk * 32 + hi * 16;
    v16h b;
#pragma unroll
    for (int j = 0; j < 16; ++j) b[j] = src[j];
    bw[kk] = b;
  }

  // ---- TDM prologue: stage tiles for t0 into buffer 0 (wave 0 only) ----
  if (wave == 0) {
    tdm_load_f16_1d(lds_off(&kbuf[0][0][0][0]),
                    kn_base + (unsigned long long)t0 * TILE_HALFS * 2, TILE_HALFS);
    tdm_load_f16_1d(lds_off(&vbuf[0][0][0][0]),
                    vt_base + (unsigned long long)t0 * TILE_HALFS * 2, TILE_HALFS);
  }

  for (int it = 0; it < T_TILE; ++it) {
    const int t   = t0 + it;
    const int buf = it & 1;

    // Issue next tiles into the other buffer, then counted-wait: TDM ops
    // complete in order, so TENSORcnt<=2 means the current tiles are in LDS.
    if (wave == 0) {
      if (it + 1 < T_TILE) {
        tdm_load_f16_1d(lds_off(&kbuf[buf ^ 1][0][0][0]),
                        kn_base + (unsigned long long)(t + 1) * TILE_HALFS * 2,
                        TILE_HALFS);
        tdm_load_f16_1d(lds_off(&vbuf[buf ^ 1][0][0][0]),
                        vt_base + (unsigned long long)(t + 1) * TILE_HALFS * 2,
                        TILE_HALFS);
        __builtin_amdgcn_s_wait_tensorcnt(2);
      } else {
        __builtin_amdgcn_s_wait_tensorcnt(0);
      }
    }
    __syncthreads();   // publish current K/V tiles to all waves

    // ================= Stage A: S = (qn @ kn^T) per head ==================
    v16h bk[2];
#pragma unroll
    for (int nt = 0; nt < 2; ++nt) {
      v16h b;
      {                              // k = hi*8 + j (j<8); elements 8..15 are zero-padded k>=16
        const _Float16* src = &kbuf[buf][wave][nt * 16 + ln][hi * 8];
#pragma unroll
        for (int j = 0; j < 8; ++j) b[j] = src[j];
#pragma unroll
        for (int j = 8; j < 16; ++j) b[j] = (_Float16)0.f;
      }
      bk[nt] = b;
    }
#pragma unroll
    for (int mt = 0; mt < 2; ++mt)
#pragma unroll
      for (int nt = 0; nt < 2; ++nt) {
        v8f c = {};
        c = wmma_f32_16x16x32(aq[mt], bk[nt], c);
#pragma unroll
        for (int r = 0; r < 8; ++r)
          S[wave][mt * 16 + r + 8 * hi][nt * 16 + ln] = c[r];
      }
    __syncthreads();

    // ================= Stage B: mask + scale + softmax ====================
    {
      const int h = tid >> 5;
      const int q = tid & 31;
      const unsigned char* mrow = kmask + (size_t)t * KL;
      float row[KL];
      float mx = -3.4e38f;
#pragma unroll
      for (int k = 0; k < KL; ++k) {
        float s = S[h][q][k] * 0.125f;            // 1/sqrt(64)
        if (mrow[k]) s = -3.0e38f;                // ~finfo(f32).min
        row[k] = s;
        mx = fmaxf(mx, s);
      }
      float sum = 0.f;
#pragma unroll
      for (int k = 0; k < KL; ++k) { float e = __expf(row[k] - mx); row[k] = e; sum += e; }
      const float inv = 1.f / sum;
#pragma unroll
      for (int k = 0; k < KL; ++k) P[h][q][k] = (_Float16)(row[k] * inv);
    }
    __syncthreads();

    // ================= Stage C: O_h = P @ V  (K = 32 exactly) =============
    {
      // B[k][e] = vnT[t,h,e,k]: lane holds column e=ln, halfs k = hi*16+j
      // -> contiguous 32B LDS load thanks to the transposed V layout.
      v16h bv;
      const _Float16* vsrc = &vbuf[buf][wave][ln][hi * 16];
#pragma unroll
      for (int j = 0; j < 16; ++j) bv[j] = vsrc[j];
#pragma unroll
      for (int mt = 0; mt < 2; ++mt) {
        const int m = mt * 16 + ln;
        v16h a;
#pragma unroll
        for (int j = 0; j < 8; ++j) a[j]     = P[wave][m][hi * 16 + j];
#pragma unroll
        for (int j = 0; j < 8; ++j) a[8 + j] = P[wave][m][hi * 16 + 8 + j];
        v8f c = {};
        c = wmma_f32_16x16x32(a, bv, c);
#pragma unroll
        for (int r = 0; r < 8; ++r)
          VA[mt * 16 + r + 8 * hi][wave * HD + ln] = (_Float16)c[r];
      }
    }
    __syncthreads();

    // ================= Stage D: out = VA @ W_fc^T  (K = 64) ===============
    float* dst = out + ((size_t)(i * T_DIM + t) * QL) * DM;
#pragma unroll
    for (int mt = 0; mt < 2; ++mt) {
      v8f acc = {};
#pragma unroll
      for (int kk = 0; kk < 2; ++kk) {
        const int m = mt * 16 + ln;
        v16h a;
#pragma unroll
        for (int j = 0; j < 8; ++j) a[j]     = VA[m][kk * 32 + hi * 16 + j];
#pragma unroll
        for (int j = 0; j < 8; ++j) a[8 + j] = VA[m][kk * 32 + hi * 16 + 8 + j];
        acc = wmma_f32_16x16x32(a, bw[kk], acc);
      }
#pragma unroll
      for (int r = 0; r < 8; ++r)
        dst[(mt * 16 + r + 8 * hi) * DM + wave * 16 + ln] = acc[r];
    }
    __syncthreads();   // VA / S reuse fence before next iteration
  }
}

// ---------------------------------------------------------------------------
extern "C" void kernel_launch(void* const* d_in, const int* in_sizes, int n_in,
                              void* d_out, int out_size, void* d_ws, size_t ws_size,
                              hipStream_t stream) {
  const float* query = (const float*)d_in[0];
  const float* key   = (const float*)d_in[1];
  const unsigned char* kmask = (const unsigned char*)d_in[2];  // jnp.bool_ (1B)
  const float* W_Q   = (const float*)d_in[3];
  const float* W_K   = (const float*)d_in[4];
  const float* W_V   = (const float*)d_in[5];
  const float* W_fc  = (const float*)d_in[6];
  float* out = (float*)d_out;

  // workspace layout (f16): qn | kn | vnT | wfc
  const size_t NQ = (size_t)I_DIM * HEADS * QL * HD;           // 262144
  char* ws = (char*)d_ws;
  _Float16* qn  = (_Float16*)(ws);
  _Float16* kn  = (_Float16*)(ws + NQ * sizeof(_Float16));
  _Float16* vnT = (_Float16*)(ws + 2 * NQ * sizeof(_Float16));
  _Float16* wfc = (_Float16*)(ws + 3 * NQ * sizeof(_Float16));

  {
    const int total = (int)(3 * NQ + DM * DM);
    const int threads = 256;
    const int blocks = (total + threads - 1) / threads;
    attn_proj_f16_kernel<<<blocks, threads, 0, stream>>>(
        query, key, W_Q, W_K, W_V, W_fc, qn, kn, vnT, wfc);
  }
  {
    const int blocks = I_DIM * (T_DIM / T_TILE);   // 2048
    attn_main_wmma_kernel<<<blocks, 128, 0, stream>>>(qn, kn, vnT, wfc, kmask, out);
  }
}
